// MultiHeadAttention_57853209477577
// MI455X (gfx1250) — matrix operand order repaired and checked by reading, not compiled, not run on hardware
//
#include <hip/hip_runtime.h>

typedef __bf16 bf16;
typedef __bf16 v16bf __attribute__((ext_vector_type(16)));
typedef __bf16 v8bf  __attribute__((ext_vector_type(8)));
typedef float  v8f   __attribute__((ext_vector_type(8)));

#define D_MODEL 1024
#define NHEAD   16
#define DK      64
#define BATCH   4
#define SEQ     2048
#define MROWS   (BATCH*SEQ)   // 8192

// softmax scale folded into Q at projection time: 1/sqrt(DK) * log2(e)
#define QSCALE  (0.125f * 1.44269504088896340736f)

// ---------------------------------------------------------------------------
// WMMA helper: D = A(16x32 bf16) * B(32x16 bf16) + C(16x16 f32)
// ---------------------------------------------------------------------------
__device__ __forceinline__ v8f wmma_bf16(v16bf a, v16bf b, v8f c) {
  return __builtin_amdgcn_wmma_f32_16x16x32_bf16(
      /*neg_a=*/false, a, /*neg_b=*/false, b,
      /*c_mod=*/(short)0, c, /*reuse_a=*/false, /*reuse_b=*/false);
}

__device__ __forceinline__ v8f zero8() {
  v8f z;
#pragma unroll
  for (int i = 0; i < 8; ++i) z[i] = 0.0f;
  return z;
}

__device__ __forceinline__ v16bf cat8(v8bf c0, v8bf c1) {
  return __builtin_shufflevector(c0, c1, 0,1,2,3,4,5,6,7,8,9,10,11,12,13,14,15);
}

// A-fragment (contiguous k order): per lane 16 contiguous elements
// at p[0..15]; caller folds in kBase + 16*(lane>>4).
__device__ __forceinline__ v16bf load_a(const bf16* p) {
  v8bf c0 = *(const v8bf*)p;
  v8bf c1 = *(const v8bf*)(p + 8);
  return cat8(c0, c1);
}
__device__ __forceinline__ v16bf load_a(const float* p) {
  v8f x0 = *(const v8f*)p;
  v8f x1 = *(const v8f*)(p + 8);
  v8bf c0, c1;
#pragma unroll
  for (int i = 0; i < 8; ++i) { c0[i] = (bf16)x0[i]; c1[i] = (bf16)x1[i]; }
  return cat8(c0, c1);
}

// ---------------------------------------------------------------------------
// fp32 -> bf16 conversion (weights), 4 elements/thread
// ---------------------------------------------------------------------------
__global__ void cvt_f32_to_bf16(const float* __restrict__ in,
                                bf16* __restrict__ out, int n4) {
  int i = (blockIdx.x * blockDim.x + threadIdx.x) * 4;
  if (i < n4 * 4) {
    float4 x = *(const float4*)(in + i);
    out[i + 0] = (bf16)x.x;
    out[i + 1] = (bf16)x.y;
    out[i + 2] = (bf16)x.z;
    out[i + 3] = (bf16)x.w;
  }
}

// ---------------------------------------------------------------------------
// GEMM: Y = (X @ W^T + bias) * oscale.  X: [8192,1024] (f32 or bf16),
// W: [N,K] bf16.  Each wave: 64(M) x 64(N) tile, K step 32 -> 16 WMMAs/step.
// MODE 0: store bf16 to [B,H,S,DK]   (Q with oscale=QSCALE, K with 1.0)
// MODE 1: store bf16 to [B,H,DK,S]   (V transposed for P@V B-fragments)
// MODE 2: store f32 row-major to d_out (output projection)
// ---------------------------------------------------------------------------
template <typename AT, int MODE>
__global__ __launch_bounds__(128) void gemm_xwT(
    const AT* __restrict__ A, const bf16* __restrict__ W,
    const float* __restrict__ bias, void* __restrict__ outp, float oscale) {
  const int lane   = threadIdx.x & 31;
  const int wave   = threadIdx.x >> 5;
  const int lanelo = lane & 15;
  const int lhi    = lane >> 4;
  const int m0 = (blockIdx.x * 4 + wave) * 64;
  const int n0 = blockIdx.y * 64;

  v8f acc[4][4];
#pragma unroll
  for (int mt = 0; mt < 4; ++mt)
#pragma unroll
    for (int nt = 0; nt < 4; ++nt) acc[mt][nt] = zero8();

  const AT* arow[4];
#pragma unroll
  for (int mt = 0; mt < 4; ++mt)
    arow[mt] = A + (size_t)(m0 + mt * 16 + lanelo) * D_MODEL;

  for (int kk = 0; kk < D_MODEL; kk += 32) {
    v16bf a[4];
#pragma unroll
    for (int mt = 0; mt < 4; ++mt) a[mt] = load_a(arow[mt] + kk + 16 * lhi);
#pragma unroll
    for (int nt = 0; nt < 4; ++nt) {
      // B[k][n] = W[n][k]: lane = column n, 16 contiguous k at 16*(lane>>4)
      v16bf bw = *(const v16bf*)(W + (size_t)(n0 + nt * 16 + lanelo) * D_MODEL
                                 + kk + 16 * lhi);
#pragma unroll
      for (int mt = 0; mt < 4; ++mt)
        acc[mt][nt] = wmma_bf16(a[mt], bw, acc[mt][nt]);
    }
  }

#pragma unroll
  for (int nt = 0; nt < 4; ++nt) {
    const int n = n0 + nt * 16 + lanelo;
    const float bb = bias[n];
#pragma unroll
    for (int mt = 0; mt < 4; ++mt) {
#pragma unroll
      for (int v = 0; v < 8; ++v) {
        const float val = (acc[mt][nt][v] + bb) * oscale;
        const int m = m0 + mt * 16 + v + 8 * lhi;   // D layout: row = v+8*half
        if constexpr (MODE == 2) {
          ((float*)outp)[(size_t)m * D_MODEL + n] = val;
        } else {
          const int b = m >> 11, s = m & (SEQ - 1);
          const int h = n >> 6,  d = n & (DK - 1);
          size_t idx;
          if constexpr (MODE == 0)
            idx = (((size_t)(b * NHEAD + h)) * SEQ + s) * DK + d;
          else
            idx = (((size_t)(b * NHEAD + h)) * DK + d) * SEQ + s;
          ((bf16*)outp)[idx] = (bf16)val;
        }
      }
    }
  }
}

// ---------------------------------------------------------------------------
// Flash attention, one wave = 16 query rows, key blocks of 64.
// Q (pre-scaled by QSCALE), K: [B,H,S,DK] bf16; Vt: [B,H,DK,S] bf16;
// O: [B,S,H*DK] bf16.
// ---------------------------------------------------------------------------
__device__ __forceinline__ float red_max16(float x) {
  x = fmaxf(x, __shfl_xor(x, 1, 32));
  x = fmaxf(x, __shfl_xor(x, 2, 32));
  x = fmaxf(x, __shfl_xor(x, 4, 32));
  x = fmaxf(x, __shfl_xor(x, 8, 32));   // masks <16 keep lane halves separate
  return x;
}
__device__ __forceinline__ float red_sum16(float x) {
  x += __shfl_xor(x, 1, 32);
  x += __shfl_xor(x, 2, 32);
  x += __shfl_xor(x, 4, 32);
  x += __shfl_xor(x, 8, 32);
  return x;
}

__global__ __launch_bounds__(128) void flash_attn_kernel(
    const bf16* __restrict__ Q, const bf16* __restrict__ K,
    const bf16* __restrict__ Vt, bf16* __restrict__ O) {
  alignas(64) __shared__ bf16 ldsP[4][16][64];   // per-wave P staging (2 KB)

  const int lane   = threadIdx.x & 31;
  const int wave   = threadIdx.x >> 5;
  const int lanelo = lane & 15;
  const int lhi    = lane >> 4;
  const int h = blockIdx.y, b = blockIdx.z;
  const int q0 = (blockIdx.x * 4 + wave) * 16;

  const size_t bh = (size_t)(b * NHEAD + h);
  const bf16* Qbh = Q  + bh * SEQ * DK;
  const bf16* Kbh = K  + bh * SEQ * DK;
  const bf16* Vbh = Vt + bh * DK * SEQ;

  // Q tile 16x64 -> two A-fragments (d=0..31, d=32..63), loaded once
  const bf16* qrow = Qbh + (size_t)(q0 + lanelo) * DK;
  const v16bf aq0 = load_a(qrow + 16 * lhi);
  const v16bf aq1 = load_a(qrow + 32 + 16 * lhi);

  float mrow[8], lrow[8];
  v8f acc[4];
#pragma unroll
  for (int v = 0; v < 8; ++v) { mrow[v] = -1e30f; lrow[v] = 0.0f; }
#pragma unroll
  for (int dt = 0; dt < 4; ++dt) acc[dt] = zero8();

  for (int kb = 0; kb < SEQ; kb += 64) {
    // ---- scores (already in log2 domain: Q pre-scaled by QSCALE) ----
    v8f t[4];
#pragma unroll
    for (int j = 0; j < 4; ++j) {
      const bf16* kr = Kbh + (size_t)(kb + j * 16 + lanelo) * DK;
      v8f s = zero8();
      s = wmma_bf16(aq0, *(const v16bf*)(kr + 16 * lhi), s);
      s = wmma_bf16(aq1, *(const v16bf*)(kr + 32 + 16 * lhi), s);
      t[j] = s;
    }

    // ---- online softmax (row stats via 16-lane xor reductions) ----
    float alpha[8];
#pragma unroll
    for (int v = 0; v < 8; ++v) {
      float x = fmaxf(fmaxf(t[0][v], t[1][v]), fmaxf(t[2][v], t[3][v]));
      x = red_max16(x);
      const float mn = fmaxf(mrow[v], x);
      const float a  = exp2f(mrow[v] - mn);
      const float p0 = exp2f(t[0][v] - mn);
      const float p1 = exp2f(t[1][v] - mn);
      const float p2 = exp2f(t[2][v] - mn);
      const float p3 = exp2f(t[3][v] - mn);
      t[0][v] = p0; t[1][v] = p1; t[2][v] = p2; t[3][v] = p3;
      lrow[v] = lrow[v] * a + red_sum16((p0 + p1) + (p2 + p3));
      mrow[v] = mn;
      alpha[v] = a;
    }
#pragma unroll
    for (int dt = 0; dt < 4; ++dt)
#pragma unroll
      for (int v = 0; v < 8; ++v) acc[dt][v] *= alpha[v];

    // ---- transpose P: D-layout -> A-layout through LDS ----
#pragma unroll
    for (int v = 0; v < 8; ++v)
#pragma unroll
      for (int j = 0; j < 4; ++j)
        ldsP[wave][v + 8 * lhi][j * 16 + lanelo] = (bf16)t[j][v];
    __builtin_amdgcn_wave_barrier();          // same-wave, in-order DS pipe
    const bf16* prow = &ldsP[wave][lanelo][0];
    v16bf ap0 = cat8(*(const v8bf*)(prow + 16 * lhi),
                     *(const v8bf*)(prow + 16 * lhi + 8));
    v16bf ap1 = cat8(*(const v8bf*)(prow + 32 + 16 * lhi),
                     *(const v8bf*)(prow + 32 + 16 * lhi + 8));
    __builtin_amdgcn_wave_barrier();

    // ---- acc += P(16x64) @ V(64x64): B from transposed V, contiguous ----
#pragma unroll
    for (int dt = 0; dt < 4; ++dt) {
      const bf16* vr = Vbh + (size_t)(dt * 16 + lanelo) * SEQ + kb;
      acc[dt] = wmma_bf16(ap0, *(const v16bf*)(vr + 16 * lhi), acc[dt]);
      acc[dt] = wmma_bf16(ap1, *(const v16bf*)(vr + 32 + 16 * lhi), acc[dt]);
    }
  }

#pragma unroll
  for (int v = 0; v < 8; ++v) lrow[v] = 1.0f / lrow[v];
#pragma unroll
  for (int dt = 0; dt < 4; ++dt) {
#pragma unroll
    for (int v = 0; v < 8; ++v) {
      const int m = q0 + v + 8 * lhi;
      const int n = h * DK + dt * 16 + lanelo;
      O[((size_t)(b * SEQ + m)) * D_MODEL + n] = (bf16)(acc[dt][v] * lrow[v]);
    }
  }
}

// ---------------------------------------------------------------------------
// launch
// ---------------------------------------------------------------------------
extern "C" void kernel_launch(void* const* d_in, const int* in_sizes, int n_in,
                              void* d_out, int out_size, void* d_ws,
                              size_t ws_size, hipStream_t stream) {
  const float* q  = (const float*)d_in[0];
  const float* k  = (const float*)d_in[1];
  const float* v  = (const float*)d_in[2];
  const float* Wq = (const float*)d_in[3];
  const float* bq = (const float*)d_in[4];
  const float* Wk = (const float*)d_in[5];
  const float* bk = (const float*)d_in[6];
  const float* Wv = (const float*)d_in[7];
  const float* bv = (const float*)d_in[8];
  const float* Wo = (const float*)d_in[9];
  const float* bo = (const float*)d_in[10];

  // workspace layout (bf16 elements): 4 weights (4M) + Q (8M) + K (8M)
  // + Vt (8M) + attention output (8M) = 72 MB
  bf16* ws = (bf16*)d_ws;
  const size_t WSZ = (size_t)D_MODEL * D_MODEL;
  bf16* Wqb = ws;
  bf16* Wkb = ws + WSZ;
  bf16* Wvb = ws + 2 * WSZ;
  bf16* Wob = ws + 3 * WSZ;
  bf16* Qb  = ws + 4 * WSZ;
  bf16* Kb  = Qb + (size_t)MROWS * D_MODEL;
  bf16* Vtb = Kb + (size_t)MROWS * D_MODEL;
  bf16* Ab  = Vtb + (size_t)MROWS * D_MODEL;

  {
    const int n4 = (int)(WSZ / 4);
    dim3 grid((unsigned)((n4 + 255) / 256)), blk(256);
    cvt_f32_to_bf16<<<grid, blk, 0, stream>>>(Wq, Wqb, n4);
    cvt_f32_to_bf16<<<grid, blk, 0, stream>>>(Wk, Wkb, n4);
    cvt_f32_to_bf16<<<grid, blk, 0, stream>>>(Wv, Wvb, n4);
    cvt_f32_to_bf16<<<grid, blk, 0, stream>>>(Wo, Wob, n4);
  }

  dim3 gg(MROWS / 256, D_MODEL / 64), gb(128);
  gemm_xwT<float, 0><<<gg, gb, 0, stream>>>(q, Wqb, bq, Qb, QSCALE);
  gemm_xwT<float, 0><<<gg, gb, 0, stream>>>(k, Wkb, bk, Kb, 1.0f);
  gemm_xwT<float, 1><<<gg, gb, 0, stream>>>(v, Wvb, bv, Vtb, 1.0f);

  dim3 ag(SEQ / 64, NHEAD, BATCH);
  flash_attn_kernel<<<ag, gb, 0, stream>>>(Qb, Kb, Vtb, Ab);

  gemm_xwT<bf16, 2><<<gg, gb, 0, stream>>>(Ab, Wob, bo, d_out, 1.0f);
}
